// MultiScalePeriodDetector_58110907515013
// MI455X (gfx1250) — matrix operand order repaired and checked by reading, not compiled, not run on hardware
//
#include <hip/hip_runtime.h>
#include <hip/hip_bf16.h>
#include <math.h>

// ---------------------------------------------------------------------------
// MultiScalePeriodDetector for MI455X (gfx1250, wave32, WMMA)
//   FFT-4096 = 64x64 four-step algorithm; 64-pt DFT stages are complex GEMMs
//   executed with v_wmma_f32_16x16x32_f16 (f16 in, f32 accumulate).
//   A-side fragments are hoisted out of the N-tile loop (built once per wave),
//   and the source-layout switch is a template parameter (no hot-loop branch).
//   VMD runs as one workgroup per sequence; the final ifft is eliminated by
//   computing IMF energies / spectral argmax directly from u_hat.
// ---------------------------------------------------------------------------

typedef _Float16 v16h __attribute__((ext_vector_type(16)));
typedef float    v8f  __attribute__((ext_vector_type(8)));

#define N_SEQ   1024
#define T_LEN   4096
#define RBINS   2049      // rfft bins for T=4096
#define PMAX    8200      // period count table size (max period = 8192)
#define ALPHA_C 2000.0f
#define TAU_C   1e-7f
#define TAU2_C  1e-14f

__device__ __forceinline__ v8f wmma_f16(v16h a, v16h b, v8f c) {
  // D = A(16x32 f16) * B(32x16 f16) + C(16x16 f32)
  return __builtin_amdgcn_wmma_f32_16x16x32_f16(false, a, false, b, (short)0, c,
                                                false, false);
}

__device__ __forceinline__ int imax_i(int a, int b) { return a > b ? a : b; }
__device__ __forceinline__ int imin_i(int a, int b) { return a < b ? a : b; }

// ISA 16-bit A 16x32 layout: K index for fragment half j (0..15)
__device__ __forceinline__ int a_kidx(int j, int lane, int kt) {
  const int v = j >> 1, lo = j & 1;
  return kt * 32 + ((v < 4) ? 0 : 16) + ((lane < 16) ? 0 : 8) + 2 * (v & 3) + lo;
}
// ISA 16-bit B 32x16 layout: K index for fragment half j (0..15)
__device__ __forceinline__ int b_kidx(int j, int lane, int kt) {
  const int v = j >> 1, lo = j & 1;
  return kt * 32 + ((v < 4) ? 0 : 16) + ((lane < 16) ? 0 : 8) + 2 * (v & 3) + lo;
}

// ---------------------------------------------------------------------------
// Stage 1:  T1[k1][n2] = sum_n1 F64[k1][n1] * S[n1][n2]   (S real)
// One wave per (seq, mtile).  SRCMODE 0: gather from x[B,T,C]; 1: [N][4096].
// ---------------------------------------------------------------------------
template <int SRCMODE>
__global__ __launch_bounds__(256) void dft_stage1(const float* __restrict__ src,
                                                  float* __restrict__ outR,
                                                  float* __restrict__ outI) {
  const int wid   = (blockIdx.x * blockDim.x + threadIdx.x) >> 5;
  const int lane  = threadIdx.x & 31;
  const int seq   = wid >> 2;
  const int mtile = wid & 3;
  const int m     = mtile * 16 + (lane & 15);   // output row (k1)
  const int laneN = lane & 15;
  const int bx = seq >> 5, cx = seq & 31;

  // A fragments (F64) depend only on (m, kt): build once per wave.
  v16h arF[2], aiF[2];
#pragma unroll
  for (int kt = 0; kt < 2; ++kt) {
#pragma unroll
    for (int j = 0; j < 16; ++j) {
      const int Ka = a_kidx(j, lane, kt);
      const int ph = (m * Ka) & 63;
      const float th = (float)ph * 0.09817477042f;   // 2*pi/64
      arF[kt][j] = (_Float16)__cosf(th);
      aiF[kt][j] = (_Float16)(-__sinf(th));
    }
  }

  for (int ntile = 0; ntile < 4; ++ntile) {
    v8f accR = {0.f,0.f,0.f,0.f,0.f,0.f,0.f,0.f};
    v8f accI = {0.f,0.f,0.f,0.f,0.f,0.f,0.f,0.f};
    const int n2 = ntile * 16 + laneN;          // output col (n2)
#pragma unroll
    for (int kt = 0; kt < 2; ++kt) {
      v16h bf;
#pragma unroll
      for (int j = 0; j < 16; ++j) {
        const int Kb = b_kidx(j, lane, kt);
        const int t  = 64 * Kb + n2;
        const float sv = (SRCMODE == 0)
                             ? src[((size_t)bx * T_LEN + t) * 32 + cx]
                             : src[(size_t)seq * T_LEN + t];
        bf[j] = (_Float16)sv;
      }
      accR = wmma_f16(arF[kt], bf, accR);
      accI = wmma_f16(aiF[kt], bf, accI);
    }
#pragma unroll
    for (int r = 0; r < 8; ++r) {
      const int M = mtile * 16 + r + ((lane < 16) ? 0 : 8);
      const int N = ntile * 16 + laneN;
      outR[(size_t)seq * T_LEN + M * 64 + N] = accR[r];
      outI[(size_t)seq * T_LEN + M * 64 + N] = accI[r];
    }
  }
}

// ---------------------------------------------------------------------------
// Stage 2:  X[k1][k2] = sum_n2 (T1[k1][n2]*W4096^{k1*n2}) * F64[n2][k2]
// Complex x complex -> 4 real WMMA GEMMs per k-step.  A fragments (twiddled
// stage-1 tile) are loaded + twiddled once per wave, reused for all 4 N-tiles.
// ---------------------------------------------------------------------------
__global__ __launch_bounds__(256) void dft_stage2(const float* __restrict__ inR,
                                                  const float* __restrict__ inI,
                                                  float* __restrict__ outR,
                                                  float* __restrict__ outI) {
  const int wid   = (blockIdx.x * blockDim.x + threadIdx.x) >> 5;
  const int lane  = threadIdx.x & 31;
  const int seq   = wid >> 2;
  const int mtile = wid & 3;
  const int m     = mtile * 16 + (lane & 15);   // k1
  const int laneN = lane & 15;

  // A fragments depend only on (m, kt): load + twiddle once per wave.
  v16h ArF[2], AiF[2];
#pragma unroll
  for (int kt = 0; kt < 2; ++kt) {
#pragma unroll
    for (int j = 0; j < 16; ++j) {
      const int Ka = a_kidx(j, lane, kt);       // n2
      const float tr = inR[(size_t)seq * T_LEN + m * 64 + Ka];
      const float ti = inI[(size_t)seq * T_LEN + m * 64 + Ka];
      const float th = (float)(m * Ka) * 0.001533980788f;  // 2*pi/4096
      const float cc = __cosf(th), ss = __sinf(th);
      ArF[kt][j] = (_Float16)(tr * cc + ti * ss);   // (tr+i*ti)*(cc - i*ss)
      AiF[kt][j] = (_Float16)(ti * cc - tr * ss);
    }
  }

  for (int ntile = 0; ntile < 4; ++ntile) {
    v8f accR = {0.f,0.f,0.f,0.f,0.f,0.f,0.f,0.f};
    v8f accI = {0.f,0.f,0.f,0.f,0.f,0.f,0.f,0.f};
    const int n = ntile * 16 + laneN;           // k2
#pragma unroll
    for (int kt = 0; kt < 2; ++kt) {
      v16h Fr, Fi, nFi;
#pragma unroll
      for (int j = 0; j < 16; ++j) {
        const int Kb = b_kidx(j, lane, kt);     // n2
        const int ph = (Kb * n) & 63;
        const float t2 = (float)ph * 0.09817477042f;
        const float fc = __cosf(t2), fs = __sinf(t2);
        Fr[j]  = (_Float16)fc;
        Fi[j]  = (_Float16)(-fs);
        nFi[j] = (_Float16)fs;                  // -Fi (f16 WMMA has no A/B neg)
      }
      accR = wmma_f16(ArF[kt], Fr, accR);
      accR = wmma_f16(AiF[kt], nFi, accR);
      accI = wmma_f16(ArF[kt], Fi, accI);
      accI = wmma_f16(AiF[kt], Fr, accI);
    }
#pragma unroll
    for (int r = 0; r < 8; ++r) {
      const int M = mtile * 16 + r + ((lane < 16) ? 0 : 8);
      const int N = ntile * 16 + laneN;
      outR[(size_t)seq * T_LEN + M * 64 + N] = accR[r];
      outI[(size_t)seq * T_LEN + M * 64 + N] = accI[r];
    }
  }
}

// ---------------------------------------------------------------------------
// Accumulate |X| into per-frequency (all seqs) and per-B sums.
// Bin f lives at tile index (f%64)*64 + f/64.
// ---------------------------------------------------------------------------
__global__ void fft_amp_accum(const float* __restrict__ Xr,
                              const float* __restrict__ Xi,
                              float* __restrict__ freqSum,
                              float* __restrict__ perB) {
  const int gid = blockIdx.x * blockDim.x + threadIdx.x;
  if (gid >= 32 * RBINS) return;
  const int b = gid / RBINS, f = gid % RBINS;
  if (f == 0) { perB[gid] = 0.f; return; }
  const int idx = (f & 63) * 64 + (f >> 6);
  float s = 0.f;
  for (int c = 0; c < 32; ++c) {
    const size_t base = (size_t)(b * 32 + c) * T_LEN + idx;
    __builtin_prefetch(&Xr[base], 0, 1);  // global_prefetch_b8
    const float r = Xr[base], im = Xi[base];
    s += sqrtf(r * r + im * im);
  }
  perB[gid] = s;
  atomicAdd(&freqSum[f], s);
}

// ---------------------------------------------------------------------------
// db4 single-level DWT (symmetric pad), cA zero-padded into [N][4096].
// ---------------------------------------------------------------------------
__global__ __launch_bounds__(256) void dwt_db4(const float* __restrict__ x,
                                               float* __restrict__ bufA,
                                               float* __restrict__ sumWavE) {
  const float DB4[8] = {-0.010597401784997278f, 0.032883011666982945f,
                        0.030841381835986965f,  -0.18703481171888114f,
                        -0.02798376941698385f,  0.6308807679295904f,
                        0.7148465705525415f,    0.23037781330885523f};
  const int seq = blockIdx.x, tid = threadIdx.x;
  const int b = seq >> 5, c = seq & 31;
  float e = 0.f;
  for (int i = tid; i < T_LEN; i += 256) {
    float v = 0.f;
    if (i < 2051) {
#pragma unroll
      for (int j = 0; j < 8; ++j) {
        int q = 2 * i + 1 - j;                 // cA[i] = conv[2i+1]
        if (q < 0) q = -q - 1;                 // symmetric left pad
        if (q >= T_LEN) q = 2 * T_LEN - 1 - q; // symmetric right pad
        v += x[((size_t)b * T_LEN + q) * 32 + c] * DB4[j];
      }
      e += v * v;
    }
    bufA[(size_t)seq * T_LEN + i] = v;         // zero-pad tail
  }
  __shared__ float red[256];
  red[tid] = e; __syncthreads();
  for (int s = 128; s > 0; s >>= 1) {
    if (tid < s) red[tid] += red[tid + s];
    __syncthreads();
  }
  if (tid == 0) atomicAdd(sumWavE, red[0]);
}

// ---------------------------------------------------------------------------
// Per-sequence top-5 wavelet spectrum bins -> period candidates (counts).
// ---------------------------------------------------------------------------
__global__ __launch_bounds__(256) void wav_top5(const float* __restrict__ Wr,
                                                const float* __restrict__ Wi,
                                                unsigned* __restrict__ countWav) {
  __shared__ float amp[RBINS];
  __shared__ float rv[256];
  __shared__ int   ri[256];
  const int seq = blockIdx.x, tid = threadIdx.x;
  for (int f = tid; f < RBINS; f += 256) {
    if (f == 0) { amp[0] = 0.f; continue; }
    const int idx = (f & 63) * 64 + (f >> 6);
    const float r = Wr[(size_t)seq * T_LEN + idx];
    const float im = Wi[(size_t)seq * T_LEN + idx];
    amp[f] = sqrtf(r * r + im * im);
  }
  __syncthreads();
  for (int round = 0; round < 5; ++round) {
    float bv = -1.f; int bi = 1;
    for (int f = tid; f < RBINS; f += 256)
      if (amp[f] > bv) { bv = amp[f]; bi = f; }
    rv[tid] = bv; ri[tid] = bi; __syncthreads();
    for (int s = 128; s > 0; s >>= 1) {
      if (tid < s && rv[tid + s] > rv[tid]) { rv[tid] = rv[tid + s]; ri[tid] = ri[tid + s]; }
      __syncthreads();
    }
    if (tid == 0) {
      const int idx = ri[0];
      amp[idx] = -2.f;
      const int p = 2 * (T_LEN / imax_i(idx, 1));  // rescaled for zero-padding
      atomicAdd(&countWav[imin_i(p, PMAX - 1)], 1u);
    }
    __syncthreads();
  }
}

// ---------------------------------------------------------------------------
// VMD: one workgroup per sequence.  f_hat = fftshift(X).  49 full ADMM
// updates, then a final u-pass (omega_49, lambda_49) per mode with
// frequency-domain energy + rfft-amplitude argmax (no ifft needed).
// ---------------------------------------------------------------------------
__device__ __forceinline__ float blockReduceSum256(float v, float* red, int tid) {
  red[tid] = v; __syncthreads();
  for (int s = 128; s > 0; s >>= 1) {
    if (tid < s) red[tid] += red[tid + s];
    __syncthreads();
  }
  const float r = red[0]; __syncthreads();
  return r;
}

__global__ __launch_bounds__(256) void vmd_kernel(const float* __restrict__ Xr,
                                                  const float* __restrict__ Xi,
                                                  float* __restrict__ vmdE,
                                                  float* __restrict__ vmdArg) {
  __shared__ float uR[T_LEN];
  __shared__ float uI[T_LEN];
  __shared__ float red[256];
  __shared__ int   redi[256];
  const int seq = blockIdx.x, tid = threadIdx.x;

  float fr[16], fi[16], lr[16], li[16], fq[16];
#pragma unroll
  for (int i = 0; i < 16; ++i) {
    const int t = tid + 256 * i;
    const int g = (t + 2048) & 4095;             // fftshift
    const int idx = (g & 63) * 64 + (g >> 6);
    fr[i] = Xr[(size_t)seq * T_LEN + idx];
    fi[i] = Xi[(size_t)seq * T_LEN + idx];
    lr[i] = 0.f; li[i] = 0.f;
    fq[i] = (float)t * (1.0f / 4096.0f) - 0.5f;  // freqs[t] = t/T - 0.5
  }
  float w[3];
#pragma unroll
  for (int k = 0; k < 3; ++k) {                  // deterministic omega0 hash
    unsigned h = (unsigned)seq * 1664525u ^ ((unsigned)k * 1013904223u) ^ 0x9E3779B9u;
    h ^= h >> 16; h *= 0x45d9f3bu; h ^= h >> 13;
    w[k] = 0.5f * (float)(h & 0xFFFFFFu) * (1.0f / 16777216.0f);
  }

  for (int iter = 0; iter < 49; ++iter) {
    float p0 = 0.f, p1 = 0.f, p2 = 0.f, q0 = 0.f, q1 = 0.f, q2 = 0.f;
#pragma unroll
    for (int i = 0; i < 16; ++i) {
      const float f = fq[i];
      const float d0 = (f - w[0]) * (f - w[0]);
      const float d1 = (f - w[1]) * (f - w[1]);
      const float d2 = (f - w[2]) * (f - w[2]);
      const float e0 = 1.f + ALPHA_C * (d0 + TAU2_C + d1);
      const float e1 = 1.f + ALPHA_C * (d1 + TAU2_C + d0 + d2);
      const float e2 = 1.f + ALPHA_C * (d2 + TAU2_C + d1);
      const float nr = fr[i] - 0.5f * lr[i];
      const float ni = fi[i] - 0.5f * li[i];
      const float u0r = nr / e0, u0i = ni / e0;
      const float u1r = nr / e1, u1i = ni / e1;
      const float u2r = nr / e2, u2i = ni / e2;
      const float m0 = u0r * u0r + u0i * u0i;
      const float m1 = u1r * u1r + u1i * u1i;
      const float m2 = u2r * u2r + u2i * u2i;
      p0 += m0; q0 += m0 * f;
      p1 += m1; q1 += m1 * f;
      p2 += m2; q2 += m2 * f;
      lr[i] += TAU_C * ((u0r + u1r + u2r) - fr[i]);
      li[i] += TAU_C * ((u0i + u1i + u2i) - fi[i]);
    }
    p0 = blockReduceSum256(p0, red, tid); q0 = blockReduceSum256(q0, red, tid);
    p1 = blockReduceSum256(p1, red, tid); q1 = blockReduceSum256(q1, red, tid);
    p2 = blockReduceSum256(p2, red, tid); q2 = blockReduceSum256(q2, red, tid);
    w[0] = q0 / p0; w[1] = q1 / p1; w[2] = q2 / p2;   // block-uniform
  }

  // Final u per mode (omega_49, lambda_49); spectral energy & argmax.
  for (int k = 0; k < 3; ++k) {
#pragma unroll
    for (int i = 0; i < 16; ++i) {
      const int t = tid + 256 * i;
      const float f = fq[i];
      const float d0 = (f - w[0]) * (f - w[0]);
      const float d1 = (f - w[1]) * (f - w[1]);
      const float d2 = (f - w[2]) * (f - w[2]);
      float den;
      if (k == 0)      den = 1.f + ALPHA_C * (d0 + TAU2_C + d1);
      else if (k == 1) den = 1.f + ALPHA_C * (d1 + TAU2_C + d0 + d2);
      else             den = 1.f + ALPHA_C * (d2 + TAU2_C + d1);
      uR[t] = (fr[i] - 0.5f * lr[i]) / den;
      uI[t] = (fi[i] - 0.5f * li[i]) / den;
    }
    __syncthreads();
    // U'[f] = u[(f+2048)&4095] (ifftshift). rfft(Re(ifft(U')))[f] =
    // (U'[f] + conj(U'[(T-f)%T]))/2 ; Parseval gives the IMF energy.
    float esum = 0.f, bv = -1.f; int bi = 1;
    for (int f2 = tid; f2 < T_LEN; f2 += 256) {
      const int a  = (f2 + 2048) & 4095;
      const int b2 = (((T_LEN - f2) & 4095) + 2048) & 4095;
      const float ar2 = uR[a] + uR[b2];
      const float ai2 = uI[a] - uI[b2];
      const float val = ar2 * ar2 + ai2 * ai2;
      esum += val;
      if (f2 >= 1 && f2 <= 2048 && val > bv) { bv = val; bi = f2; }
    }
    const float etot = blockReduceSum256(esum, red, tid) * (0.25f / (float)T_LEN);
    red[tid] = bv; redi[tid] = bi; __syncthreads();
    for (int s = 128; s > 0; s >>= 1) {
      if (tid < s && red[tid + s] > red[tid]) { red[tid] = red[tid + s]; redi[tid] = redi[tid + s]; }
      __syncthreads();
    }
    if (tid == 0) {
      vmdE[seq * 3 + k]   = etot;
      vmdArg[seq * 3 + k] = (float)redi[0];
    }
    __syncthreads();
  }
}

// ---------------------------------------------------------------------------
// Per-sequence top-2 IMFs -> period candidates + energy sum.
// ---------------------------------------------------------------------------
__global__ void vmd_cands(const float* __restrict__ vmdE,
                          const float* __restrict__ vmdArg,
                          unsigned* __restrict__ countVmd,
                          float* __restrict__ sumVmdE) {
  const int seq = blockIdx.x * blockDim.x + threadIdx.x;
  if (seq >= N_SEQ) return;
  const float e0 = vmdE[seq * 3 + 0], e1 = vmdE[seq * 3 + 1], e2 = vmdE[seq * 3 + 2];
  int i0, i1;
  if (e0 >= e1 && e0 >= e2)      { i0 = 0; i1 = (e1 >= e2) ? 1 : 2; }
  else if (e1 >= e0 && e1 >= e2) { i0 = 1; i1 = (e0 >= e2) ? 0 : 2; }
  else                           { i0 = 2; i1 = (e0 >= e1) ? 0 : 1; }
  const int order[2] = {i0, i1};
  float etop = 0.f;
  for (int r = 0; r < 2; ++r) {
    const int k = order[r];
    const int bin = (int)vmdArg[seq * 3 + k];
    const int p = T_LEN / imax_i(bin, 1);
    atomicAdd(&countVmd[imin_i(p, PMAX - 1)], 1u);
    etop += vmdE[seq * 3 + k];
  }
  atomicAdd(sumVmdE, etop);
}

// ---------------------------------------------------------------------------
// Clear accumulators (must run every launch: deterministic replays).
// ---------------------------------------------------------------------------
__global__ void clear_accum(float* __restrict__ freqSum,
                            unsigned* __restrict__ cw,
                            unsigned* __restrict__ cv,
                            float* __restrict__ sums) {
  const int i = blockIdx.x * blockDim.x + threadIdx.x;
  if (i < RBINS) freqSum[i] = 0.f;
  if (i < PMAX) { cw[i] = 0u; cv[i] = 0u; }
  if (i < 2) sums[i] = 0.f;
#if defined(__has_builtin)
#if __has_builtin(__builtin_amdgcn_s_wait_tensorcnt)
  if (i == 0) __builtin_amdgcn_s_wait_tensorcnt(0);
#endif
#endif
}

// ---------------------------------------------------------------------------
// Fusion: fft top-5, wav/vmd top-5 by count, cross-method fuse + weights.
// ---------------------------------------------------------------------------
__global__ void fuse_kernel(const float* __restrict__ freqSum,
                            const float* __restrict__ perB,
                            const unsigned* __restrict__ countWav,
                            const unsigned* __restrict__ countVmd,
                            const float* __restrict__ sums,
                            float* __restrict__ out) {
  __shared__ int   fftIdx[5], fftP[5], wavP[5], vmdP[5], fusedP[5];
  __shared__ float sWavW, sVmdW;
  const int tid = threadIdx.x;

  if (tid == 0) {
    // fft: top-5 of mean amplitude (bin 0 excluded), ties -> first index
    for (int r = 0; r < 5; ++r) {
      float best = -1.f; int bi = 1;
      for (int f = 1; f < RBINS; ++f) {
        bool taken = false;
        for (int j = 0; j < r; ++j) if (fftIdx[j] == f) taken = true;
        if (!taken && freqSum[f] > best) { best = freqSum[f]; bi = f; }
      }
      fftIdx[r] = bi;
      fftP[r]   = T_LEN / imax_i(bi, 1);
    }
    // wav / vmd: top-5 by count, ties -> larger period
    for (int r = 0; r < 5; ++r) {
      unsigned best = 0; int bp = 0;
      for (int p = 2; p < PMAX; ++p) {
        bool taken = false;
        for (int j = 0; j < r; ++j) if (wavP[j] == p) taken = true;
        if (!taken && countWav[p] >= best && countWav[p] > 0) { best = countWav[p]; bp = p; }
      }
      wavP[r] = bp;
    }
    for (int r = 0; r < 5; ++r) {
      unsigned best = 0; int bp = 0;
      for (int p = 2; p < PMAX; ++p) {
        bool taken = false;
        for (int j = 0; j < r; ++j) if (vmdP[j] == p) taken = true;
        if (!taken && countVmd[p] >= best && countVmd[p] > 0) { best = countVmd[p]; bp = p; }
      }
      vmdP[r] = bp;
    }
    // fuse all 15, keep 1 < p < T/2, top-5 by count (ties -> larger p)
    int all[15];
    for (int j = 0; j < 5; ++j) { all[j] = fftP[j]; all[5 + j] = wavP[j]; all[10 + j] = vmdP[j]; }
    for (int r = 0; r < 5; ++r) {
      int bestP = 0, bestC = 0;
      for (int i = 0; i < 15; ++i) {
        const int p = all[i];
        if (p <= 1 || p >= T_LEN / 2) continue;
        bool taken = false;
        for (int j = 0; j < r; ++j) if (fusedP[j] == p) taken = true;
        if (taken) continue;
        int cnt = 0;
        for (int j = 0; j < 15; ++j) if (all[j] == p) ++cnt;
        if (cnt > bestC || (cnt == bestC && p > bestP)) { bestC = cnt; bestP = p; }
      }
      fusedP[r] = bestP;                         // 0 => pad slot
      out[r] = (float)bestP;
    }
    sWavW = (sums[0] / (float)N_SEQ) * 0.2f;     // mean energy / top_k
    sVmdW = (sums[1] / 2048.0f) * 0.2f;
  }
  __syncthreads();

  if (tid < 32) {
    const int b = tid;
    float fw[5];
    float fsum = 0.f;
    for (int j = 0; j < 5; ++j) {
      fw[j] = perB[b * RBINS + fftIdx[j]] * (1.0f / 32.0f);
      fsum += fw[j];
    }
    const float fallback = (fsum + 5.f * sWavW + 5.f * sVmdW) * (1.0f / 15.0f);
    for (int j = 0; j < 5; ++j) {
      const int p = fusedP[j];
      float wv = 0.f;
      if (p != 0) {
        float msum = 0.f; int mc = 0;
        for (int jj = 0; jj < 5; ++jj)
          if (fftP[jj] == p) { msum += fw[jj]; ++mc; break; }   // .index(): first
        bool inWav = false, inVmd = false;
        for (int jj = 0; jj < 5; ++jj) { if (wavP[jj] == p) inWav = true; if (vmdP[jj] == p) inVmd = true; }
        if (inWav) { msum += sWavW; ++mc; }
        if (inVmd) { msum += sVmdW; ++mc; }
        wv = mc ? (msum / (float)mc) : fallback;
      }
      out[5 + b * 5 + j] = wv;
    }
  }
}

// ---------------------------------------------------------------------------
extern "C" void kernel_launch(void* const* d_in, const int* in_sizes, int n_in,
                              void* d_out, int out_size, void* d_ws, size_t ws_size,
                              hipStream_t stream) {
  (void)in_sizes; (void)n_in; (void)out_size; (void)ws_size;
  const float* x = (const float*)d_in[0];
  float* out = (float*)d_out;

  char* w = (char*)d_ws;
  size_t off = 0;
  auto take = [&](size_t bytes) -> char* {
    char* p = w + off;
    off = (off + bytes + 255) & ~(size_t)255;
    return p;
  };
  const size_t SPEC = (size_t)N_SEQ * T_LEN * sizeof(float);
  float*    Xr       = (float*)take(SPEC);
  float*    Xi       = (float*)take(SPEC);
  float*    T1r      = (float*)take(SPEC);
  float*    T1i      = (float*)take(SPEC);
  float*    Wr2      = (float*)take(SPEC);
  float*    Wi2      = (float*)take(SPEC);
  float*    bufA     = (float*)take(SPEC);
  float*    freqSum  = (float*)take(RBINS * sizeof(float));
  float*    perB     = (float*)take(32 * RBINS * sizeof(float));
  unsigned* countWav = (unsigned*)take(PMAX * sizeof(unsigned));
  unsigned* countVmd = (unsigned*)take(PMAX * sizeof(unsigned));
  float*    sums     = (float*)take(2 * sizeof(float));   // [0]=wavE, [1]=vmdE
  float*    vmdE     = (float*)take(N_SEQ * 3 * sizeof(float));
  float*    vmdArg   = (float*)take(N_SEQ * 3 * sizeof(float));

  clear_accum<<<(PMAX + 255) / 256, 256, 0, stream>>>(freqSum, countWav, countVmd, sums);

  // x spectrum (also VMD's f_hat): two WMMA DFT stages
  dft_stage1<0><<<512, 256, 0, stream>>>(x, T1r, T1i);
  dft_stage2<<<512, 256, 0, stream>>>(T1r, T1i, Xr, Xi);
  fft_amp_accum<<<(32 * RBINS + 255) / 256, 256, 0, stream>>>(Xr, Xi, freqSum, perB);

  // wavelet path: DWT then WMMA DFT of zero-padded cA
  dwt_db4<<<N_SEQ, 256, 0, stream>>>(x, bufA, &sums[0]);
  dft_stage1<1><<<512, 256, 0, stream>>>(bufA, T1r, T1i);
  dft_stage2<<<512, 256, 0, stream>>>(T1r, T1i, Wr2, Wi2);
  wav_top5<<<N_SEQ, 256, 0, stream>>>(Wr2, Wi2, countWav);

  // VMD path (reuses x spectrum as f_hat)
  vmd_kernel<<<N_SEQ, 256, 0, stream>>>(Xr, Xi, vmdE, vmdArg);
  vmd_cands<<<4, 256, 0, stream>>>(vmdE, vmdArg, countVmd, &sums[1]);

  fuse_kernel<<<1, 64, 0, stream>>>(freqSum, perB, countWav, countVmd, sums, out);
}
